// MemoryEfficientAttention_2095944041025
// MI455X (gfx1250) — compile-verified, matrix operand order checked by reading, not run on hardware
//
#include <hip/hip_runtime.h>
#include <hip/hip_bf16.h>

typedef __bf16 bf16_t;
typedef __attribute__((ext_vector_type(16))) __bf16 v16bf;
typedef __attribute__((ext_vector_type(8)))  __bf16 v8bf;
typedef __attribute__((ext_vector_type(8)))  float  v8f;
typedef __attribute__((ext_vector_type(4)))  float  v4f;

__device__ __forceinline__ v16bf cat16(v8bf lo, v8bf hi) {
    return __builtin_shufflevector(lo, hi, 0,1,2,3,4,5,6,7,8,9,10,11,12,13,14,15);
}

__device__ __forceinline__ bf16_t f2bf(float x) {
    unsigned u = __builtin_bit_cast(unsigned, x);
    u += 0x7FFFu + ((u >> 16) & 1u);            // round-to-nearest-even
    unsigned short h = (unsigned short)(u >> 16);
    return __builtin_bit_cast(bf16_t, h);
}

// Async global->LDS copy, 16B per lane, tracked by ASYNCcnt.
// lds_off: byte offset in LDS (low 32 bits of generic shared pointer).
// gbyte_off: per-lane byte offset added to the 64-bit SGPR base.
__device__ __forceinline__ void async_cp_b128(unsigned lds_off, const void* sbase,
                                              unsigned gbyte_off) {
    asm volatile("global_load_async_to_lds_b128 %0, %1, %2"
                 :: "v"(lds_off), "v"(gbyte_off), "s"(sbase)
                 : "memory");
}
__device__ __forceinline__ void async_wait0() {
    asm volatile("s_wait_asynccnt 0" ::: "memory");
}

// ---------------------------------------------------------------------------
// f32 -> bf16 conversion, 8 elements / thread
// ---------------------------------------------------------------------------
__global__ void cvt_f32_bf16(const float* __restrict__ in, bf16_t* __restrict__ out, int n8) {
    int i = blockIdx.x * blockDim.x + threadIdx.x;
    if (i >= n8) return;
    v4f a = ((const v4f*)in)[2 * i + 0];
    v4f b = ((const v4f*)in)[2 * i + 1];
    v8bf o;
    o[0] = f2bf(a[0]); o[1] = f2bf(a[1]); o[2] = f2bf(a[2]); o[3] = f2bf(a[3]);
    o[4] = f2bf(b[0]); o[5] = f2bf(b[1]); o[6] = f2bf(b[2]); o[7] = f2bf(b[3]);
    ((v8bf*)out)[i] = o;
}

// ---------------------------------------------------------------------------
// GEMM: C[M,N] = (A[M,K] @ B[N,K]^T + bias[N]) * scale   (torch Linear layout)
// 256 threads = 8 waves. Block tile 128x128, K-step 64, double-buffered LDS
// filled by global_load_async_to_lds_b128. One barrier per stage. Wave 32x64.
// ---------------------------------------------------------------------------
#define GTM 128
#define GTN 128
#define GTK 64
#define GLD 72   // LDS row stride (elements): 144 bytes, 16B-aligned rows

template <bool OUTF32>
__global__ __launch_bounds__(256) void gemm_bt(
    const bf16_t* __restrict__ A, const bf16_t* __restrict__ B,
    const float* __restrict__ bias, void* __restrict__ Cout,
    int M, int N, int K, float scale)
{
    __shared__ __attribute__((aligned(16))) bf16_t As[2][GTM * GLD];
    __shared__ __attribute__((aligned(16))) bf16_t Bs[2][GTN * GLD];

    const int t     = threadIdx.x;
    const int lane  = t & 31;
    const int wave  = t >> 5;
    const int lrow  = lane & 15;
    const int lhalf = lane >> 4;
    const int bm = blockIdx.y * GTM;
    const int bn = blockIdx.x * GTN;
    const int wm = (wave >> 1) * 32;   // 4 waves along M
    const int wn = (wave & 1) * 64;    // 2 waves along N

    // one stage = 128 rows x 64 cols bf16 = 1024 chunks of 16B per matrix
    auto issue_stage = [&](int buf, int k0) {
#pragma unroll
        for (int it = 0; it < 4; ++it) {
            int c = t + it * 256;
            int r = c >> 3, co = (c & 7) << 3;
            async_cp_b128((unsigned)(size_t)&As[buf][r * GLD + co], A,
                          (unsigned)(((size_t)(bm + r) * K + k0 + co) * 2));
            async_cp_b128((unsigned)(size_t)&Bs[buf][r * GLD + co], B,
                          (unsigned)(((size_t)(bn + r) * K + k0 + co) * 2));
        }
    };

    issue_stage(0, 0);

    v8f acc[2][4] = {};
    const int nstage = K / GTK;
    for (int ks = 0; ks < nstage; ++ks) {
        const int buf = ks & 1;
        async_wait0();
        __syncthreads();   // stage `buf` resident AND all prior readers of buf done
        if (ks + 1 < nstage) issue_stage(buf ^ 1, (ks + 1) * GTK);

#pragma unroll
        for (int sub = 0; sub < 2; ++sub) {
            const int ko = sub * 32;
            v16bf afrag[2], bfrag[4];
            for (int i = 0; i < 2; ++i) {
                const bf16_t* p = &As[buf][(wm + i * 16 + lrow) * GLD + ko + lhalf * 8];
                afrag[i] = cat16(*(const v8bf*)p, *(const v8bf*)(p + 16));
            }
            for (int j = 0; j < 4; ++j) {
                const bf16_t* p = &Bs[buf][(wn + j * 16 + lrow) * GLD + ko + lhalf * 8];
                bfrag[j] = cat16(*(const v8bf*)p, *(const v8bf*)(p + 16));
            }
            for (int i = 0; i < 2; ++i)
                for (int j = 0; j < 4; ++j)
                    acc[i][j] = __builtin_amdgcn_wmma_f32_16x16x32_bf16(
                        false, afrag[i], false, bfrag[j], (short)0, acc[i][j], false, false);
        }
        // no trailing barrier: next stage's writes target the other buffer and are
        // only issued after the next top-of-loop barrier.
    }

    for (int i = 0; i < 2; ++i)
        for (int j = 0; j < 4; ++j) {
            int col = bn + wn + j * 16 + lrow;
            float bv = bias[col];
            for (int e = 0; e < 8; ++e) {
                int row = bm + wm + i * 16 + lhalf * 8 + e;
                float v = (acc[i][j][e] + bv) * scale;
                if constexpr (OUTF32)
                    ((float*)Cout)[(size_t)row * N + col] = v;
                else
                    ((bf16_t*)Cout)[(size_t)row * N + col] = f2bf(v);
            }
        }
}

// ---------------------------------------------------------------------------
// Flash attention: one (q-block=64, head) per workgroup, 4 waves x 16 q-rows.
// Q pre-scaled by 1/sqrt(64) in projection. bf16 inputs, f32 online softmax.
// K tile staged via async global->LDS; V staged transposed via ds stores.
// ---------------------------------------------------------------------------
#define AHD 64   // head dim
#define ABQ 64   // q rows per block
#define ABK 64   // k rows per iteration
#define ALD 72   // LDS row stride (elements): 144 bytes, 16B-aligned

__global__ __launch_bounds__(128) void flash_attn(
    const bf16_t* __restrict__ Q, const bf16_t* __restrict__ Km,
    const bf16_t* __restrict__ Vm, bf16_t* __restrict__ Ctx, int S)
{
    __shared__ __attribute__((aligned(16))) bf16_t Ks[ABK * ALD];   // [key][dim]
    __shared__ __attribute__((aligned(16))) bf16_t Vt[AHD * ALD];   // [dim][key]
    __shared__ __attribute__((aligned(16))) bf16_t Ps[4 * 16 * ALD];

    const int t     = threadIdx.x;
    const int lane  = t & 31;
    const int wave  = t >> 5;
    const int lrow  = lane & 15;
    const int lhalf = lane >> 4;
    const int h     = blockIdx.y;
    const int qb    = blockIdx.x * ABQ;
    const int hoff  = h * AHD;
    const size_t ld = 1024;

    // preload this wave's Q tile (16 x 64) as two A-fragments
    v16bf qf[2];
    {
        const bf16_t* qrow = Q + (size_t)(qb + wave * 16 + lrow) * ld + hoff;
        for (int s2 = 0; s2 < 2; ++s2) {
            const bf16_t* p = qrow + s2 * 32 + lhalf * 8;
            qf[s2] = cat16(*(const v8bf*)p, *(const v8bf*)(p + 16));
        }
    }

    float mrow[8], lsum[8];
    v8f o[4] = {};
    for (int i = 0; i < 8; ++i) { mrow[i] = -1e30f; lsum[i] = 0.f; }

    for (int kb = 0; kb < S; kb += ABK) {
        // K: async straight to LDS. V: load + transposed scatter to LDS.
#pragma unroll
        for (int it = 0; it < 4; ++it) {
            int c = t + it * 128;
            int r = c >> 3, co = (c & 7) << 3;
            async_cp_b128((unsigned)(size_t)&Ks[r * ALD + co], Km,
                          (unsigned)(((size_t)(kb + r) * ld + hoff + co) * 2));
            v8bf vv = *(const v8bf*)(&Vm[(size_t)(kb + r) * ld + hoff + co]);
            for (int j = 0; j < 8; ++j) Vt[(co + j) * ALD + r] = vv[j];
        }
        async_wait0();
        __syncthreads();

        // S = Q @ K^T : 16 x 64 scores in 4 accumulators
        v8f sacc[4] = {};
        for (int s2 = 0; s2 < 2; ++s2)
            for (int j = 0; j < 4; ++j) {
                const bf16_t* p = &Ks[(j * 16 + lrow) * ALD + s2 * 32 + lhalf * 8];
                v16bf kf = cat16(*(const v8bf*)p, *(const v8bf*)(p + 16));
                sacc[j] = __builtin_amdgcn_wmma_f32_16x16x32_bf16(
                    false, qf[s2], false, kf, (short)0, sacc[j], false, false);
            }

        // online softmax: rows = lhalf*8 + i, cols spread over 16 lanes x 4 tiles
        float alpha[8];
        for (int i = 0; i < 8; ++i) {
            float v = fmaxf(fmaxf(sacc[0][i], sacc[1][i]),
                            fmaxf(sacc[2][i], sacc[3][i]));
            for (int off = 8; off >= 1; off >>= 1)
                v = fmaxf(v, __shfl_xor(v, off, 32));
            float mn = fmaxf(mrow[i], v);
            alpha[i] = __expf(mrow[i] - mn);
            mrow[i] = mn;
        }

        bf16_t* pw = &Ps[wave * 16 * ALD];
        float ls[8] = {0.f, 0.f, 0.f, 0.f, 0.f, 0.f, 0.f, 0.f};
        for (int j = 0; j < 4; ++j)
            for (int i = 0; i < 8; ++i) {
                float p = __expf(sacc[j][i] - mrow[i]);
                ls[i] += p;
                pw[(lhalf * 8 + i) * ALD + j * 16 + lrow] = f2bf(p);
            }
        for (int i = 0; i < 8; ++i) {
            float v = ls[i];
            for (int off = 8; off >= 1; off >>= 1) v += __shfl_xor(v, off, 32);
            lsum[i] = alpha[i] * lsum[i] + v;
            for (int j = 0; j < 4; ++j) o[j][i] *= alpha[i];
        }

        // O += P @ V  (P from per-wave LDS as A-frags, V^T gives contiguous B-frags)
        for (int s2 = 0; s2 < 2; ++s2) {
            const bf16_t* pa = &pw[lrow * ALD + s2 * 32 + lhalf * 8];
            v16bf pf = cat16(*(const v8bf*)pa, *(const v8bf*)(pa + 16));
            for (int j = 0; j < 4; ++j) {
                const bf16_t* vb = &Vt[(j * 16 + lrow) * ALD + s2 * 32 + lhalf * 8];
                v16bf vf = cat16(*(const v8bf*)vb, *(const v8bf*)(vb + 16));
                o[j] = __builtin_amdgcn_wmma_f32_16x16x32_bf16(
                    false, pf, false, vf, (short)0, o[j], false, false);
            }
        }
        __syncthreads();   // needed: next iteration writes Ks/Vt before any barrier
    }

    for (int j = 0; j < 4; ++j)
        for (int i = 0; i < 8; ++i) {
            float v = o[j][i] / lsum[i];
            int row = qb + wave * 16 + lhalf * 8 + i;
            Ctx[(size_t)row * ld + hoff + j * 16 + lrow] = f2bf(v);
        }
}

// ---------------------------------------------------------------------------
extern "C" void kernel_launch(void* const* d_in, const int* in_sizes, int n_in,
                              void* d_out, int out_size, void* d_ws, size_t ws_size,
                              hipStream_t stream) {
    (void)in_sizes; (void)n_in; (void)out_size; (void)ws_size;
    const float* X  = (const float*)d_in[0];
    const float* Wq = (const float*)d_in[1];
    const float* bq = (const float*)d_in[2];
    const float* Wk = (const float*)d_in[3];
    const float* bk = (const float*)d_in[4];
    const float* Wv = (const float*)d_in[5];
    const float* bv = (const float*)d_in[6];
    const float* Wo = (const float*)d_in[7];
    const float* bo = (const float*)d_in[8];
    float* out = (float*)d_out;

    const int S = 4096, H = 1024;
    char* ws = (char*)d_ws;
    size_t off = 0;
    bf16_t* Xb  = (bf16_t*)(ws + off); off += (size_t)S * H * 2;
    bf16_t* Wqb = (bf16_t*)(ws + off); off += (size_t)H * H * 2;
    bf16_t* Wkb = (bf16_t*)(ws + off); off += (size_t)H * H * 2;
    bf16_t* Wvb = (bf16_t*)(ws + off); off += (size_t)H * H * 2;
    bf16_t* Wob = (bf16_t*)(ws + off); off += (size_t)H * H * 2;
    bf16_t* Qb  = (bf16_t*)(ws + off); off += (size_t)S * H * 2;
    bf16_t* Kb  = (bf16_t*)(ws + off); off += (size_t)S * H * 2;
    bf16_t* Vb  = (bf16_t*)(ws + off); off += (size_t)S * H * 2;
    bf16_t* Cb  = (bf16_t*)(ws + off); off += (size_t)S * H * 2;

    { int n8 = S * H / 8; cvt_f32_bf16<<<(n8 + 255) / 256, 256, 0, stream>>>(X,  Xb,  n8); }
    { int n8 = H * H / 8;
      cvt_f32_bf16<<<(n8 + 255) / 256, 256, 0, stream>>>(Wq, Wqb, n8);
      cvt_f32_bf16<<<(n8 + 255) / 256, 256, 0, stream>>>(Wk, Wkb, n8);
      cvt_f32_bf16<<<(n8 + 255) / 256, 256, 0, stream>>>(Wv, Wvb, n8);
      cvt_f32_bf16<<<(n8 + 255) / 256, 256, 0, stream>>>(Wo, Wob, n8); }

    dim3 gg(H / GTN, S / GTM);  // (8, 32)
    gemm_bt<false><<<gg, 256, 0, stream>>>(Xb, Wqb, bq, Qb, S, H, H, 0.125f);
    gemm_bt<false><<<gg, 256, 0, stream>>>(Xb, Wkb, bk, Kb, S, H, H, 1.0f);
    gemm_bt<false><<<gg, 256, 0, stream>>>(Xb, Wvb, bv, Vb, S, H, H, 1.0f);

    flash_attn<<<dim3(S / ABQ, 16), 128, 0, stream>>>(Qb, Kb, Vb, Cb, S);

    gemm_bt<true><<<gg, 256, 0, stream>>>(Cb, Wob, bo, out, S, H, H, 1.0f);
}
